// NCaltech101SNN_15831249453633
// MI455X (gfx1250) — matrix operand order repaired and checked by reading, not compiled, not run on hardware
//
#include <hip/hip_runtime.h>
#include <cstddef>

// ---------------------------------------------------------------------------
// MI455X (gfx1250) SNN forward.
//  Phase 1: I1_ff = x @ w1^T  -> bf16 WMMA GEMM (45 GFLOP, x read once),
//           stored transposed as [t][b][h] for streaming reads in the scan.
//  Phase 2: 300-step scan, batch split over 4 independent workgroups (M=16
//           each, zero inter-WG sync), LIF state held in WMMA-accumulator-
//           layout registers, spikes exchanged via double-buffered LDS bf16
//           (0/1 exactly representable -> spike GEMMs are exact).
// All matrix math uses v_wmma_f32_16x16x32_bf16 (wave32).
// ---------------------------------------------------------------------------

typedef __attribute__((ext_vector_type(16))) __bf16 v16bf;
typedef __attribute__((ext_vector_type(8)))  __bf16 v8bf;
typedef __attribute__((ext_vector_type(8)))  float  v8f;
typedef __attribute__((ext_vector_type(8)))  float  v8fx;   // 32B global load
typedef __attribute__((ext_vector_type(4)))  unsigned int u32x4;

// Native converts: clang emits hardware v_cvt_*_bf16 ops (RNE) on gfx1250.
static __device__ __forceinline__ __bf16 f2bf(float f) { return (__bf16)f; }
static __device__ __forceinline__ unsigned short f2bits(float f) {
    return __builtin_bit_cast(unsigned short, (__bf16)f);
}
static __device__ __forceinline__ float bf2f(unsigned short h) {
    return __builtin_bit_cast(float, (unsigned)h << 16);
}

#define WMMA_BF16(A, B, C) \
    __builtin_amdgcn_wmma_f32_16x16x32_bf16(false, (A), false, (B), (short)0, (C), false, false)

// ---------------------------------------------------------------------------
// Kernel 1: convert weights fp32 -> bf16 (w3 zero-padded to 128 rows).
// ---------------------------------------------------------------------------
__global__ __launch_bounds__(256) void convert_weights_kernel(
    const float* __restrict__ wrec, const float* __restrict__ w2,
    const float* __restrict__ w3,
    unsigned short* __restrict__ wrec_bf, unsigned short* __restrict__ w2_bf,
    unsigned short* __restrict__ w3_bf) {
    const int N1 = 512 * 512, N2 = 256 * 512, N3 = 128 * 256;
    int idx = blockIdx.x * 256 + threadIdx.x;
    if (idx < N1) {
        wrec_bf[idx] = f2bits(wrec[idx]);
    } else if (idx < N1 + N2) {
        int i = idx - N1;
        w2_bf[i] = f2bits(w2[i]);
    } else if (idx < N1 + N2 + N3) {
        int i = idx - N1 - N2;
        int r = i >> 8, c = i & 255;
        w3_bf[i] = (r < 101) ? f2bits(w3[r * 256 + c]) : (unsigned short)0;
    }
}

// ---------------------------------------------------------------------------
// Kernel 2: I1_ff[t][b][h] = sum_n x[b][t][n] * w1[h][n]
// M=19200 (= b*300+t), N=512, K=2312 (K tail zero-padded, separate iteration).
// Grid (300, 4), 256 threads = 8 waves (4 M-waves x 2 N-waves),
// wave tile = 16 x 64 (4 named WMMA accumulators).
// ---------------------------------------------------------------------------
__global__ __launch_bounds__(256) void gemm_ff_kernel(
    const float* __restrict__ x, const float* __restrict__ w1,
    float* __restrict__ I1) {
    const int K = 2312;
    const int lane = threadIdx.x & 31;
    const int wv   = threadIdx.x >> 5;
    const int mw   = wv & 3;           // 0..3
    const int nw   = wv >> 2;          // 0..1
    const int lrow = lane & 15;
    const int lhi  = lane >> 4;

    const int mbase  = blockIdx.x * 64 + mw * 16;
    const int nbase0 = blockIdx.y * 128 + nw * 64;

    const float* xrow = x + (size_t)(mbase + lrow) * K;
    const float* w1r0 = w1 + (size_t)(nbase0 + lrow) * K;

    v8f acc0 = {0.f, 0.f, 0.f, 0.f, 0.f, 0.f, 0.f, 0.f};
    v8f acc1 = acc0, acc2 = acc0, acc3 = acc0;

    const int kaOff = lhi ? 16 : 0;  // A-fragment lane K base offset
    const int kcOff = lhi ? 16 : 0;  // B-fragment lane K base offset

    // ---- main K loop: 72 full 32-wide blocks, branch-free body ----
#pragma unroll 1
    for (int kb = 0; kb < 72; ++kb) {
        const int ka = kb * 32 + kaOff;
        v8fx f0 = *(const v8fx*)(xrow + ka);
        v8fx f1 = *(const v8fx*)(xrow + ka + 8);
        v16bf afrag;
#pragma unroll
        for (int i = 0; i < 8; ++i) {
            afrag[i]     = f2bf(f0[i]);
            afrag[i + 8] = f2bf(f1[i]);
        }
        const int kc = kb * 32 + kcOff;
#pragma unroll
        for (int ti = 0; ti < 4; ++ti) {
            const float* wr = w1r0 + (size_t)(ti * 16) * K;
            v8fx g0 = *(const v8fx*)(wr + kc);
            v8fx g1 = *(const v8fx*)(wr + kc + 8);
            v16bf bfrag;
#pragma unroll
            for (int i = 0; i < 8; ++i) {
                bfrag[i]     = f2bf(g0[i]);
                bfrag[i + 8] = f2bf(g1[i]);
            }
            if      (ti == 0) acc0 = WMMA_BF16(afrag, bfrag, acc0);
            else if (ti == 1) acc1 = WMMA_BF16(afrag, bfrag, acc1);
            else if (ti == 2) acc2 = WMMA_BF16(afrag, bfrag, acc2);
            else              acc3 = WMMA_BF16(afrag, bfrag, acc3);
        }
    }
    // ---- K tail (2304..2311 valid, rest zero) ----
    {
        const int ka = 72 * 32 + kaOff;
        v16bf afrag;
#pragma unroll
        for (int i = 0; i < 8; ++i) {
            int k0 = ka + i, k1 = ka + 8 + i;
            afrag[i]     = f2bf(k0 < K ? xrow[k0] : 0.f);
            afrag[i + 8] = f2bf(k1 < K ? xrow[k1] : 0.f);
        }
        const int kc = 72 * 32 + kcOff;
#pragma unroll
        for (int ti = 0; ti < 4; ++ti) {
            const float* wr = w1r0 + (size_t)(ti * 16) * K;
            v16bf bfrag;
#pragma unroll
            for (int i = 0; i < 16; ++i) {
                int k = kc + i;
                bfrag[i] = f2bf(k < K ? wr[k] : 0.f);
            }
            if      (ti == 0) acc0 = WMMA_BF16(afrag, bfrag, acc0);
            else if (ti == 1) acc1 = WMMA_BF16(afrag, bfrag, acc1);
            else if (ti == 2) acc2 = WMMA_BF16(afrag, bfrag, acc2);
            else              acc3 = WMMA_BF16(afrag, bfrag, acc3);
        }
    }
    // ---- store transposed: row m = b*300+t  ->  I1[(t*64 + b)*512 + n] ----
    v8f* accs[4] = {&acc0, &acc1, &acc2, &acc3};
#pragma unroll
    for (int ti = 0; ti < 4; ++ti) {
        int n = nbase0 + ti * 16 + lrow;
#pragma unroll
        for (int j = 0; j < 8; ++j) {
            unsigned m = (unsigned)(mbase + j + (lhi << 3));
            unsigned b = m / 300u;
            unsigned t = m - 300u * b;
            I1[((size_t)t * 64 + b) * 512 + n] = (*accs[ti])[j];
        }
    }
}

// ---------------------------------------------------------------------------
// Helper: build 16x32 A fragment from bf16 spikes in LDS (row-major [16][KROW])
// lanes 0-15: K = kb*32 + {0..15}; lanes 16-31: +16.
// ---------------------------------------------------------------------------
static __device__ __forceinline__ v16bf lds_afrag(const unsigned short* base,
                                                  int lrow, int lhi, int kb,
                                                  int krow) {
    const unsigned short* ap = base + lrow * krow + kb * 32 + (lhi ? 16 : 0);
    v8bf lo = __builtin_bit_cast(v8bf, *(const u32x4*)ap);
    v8bf hi = __builtin_bit_cast(v8bf, *(const u32x4*)(ap + 8));
    return __builtin_shufflevector(lo, hi, 0, 1, 2, 3, 4, 5, 6, 7, 8, 9, 10,
                                   11, 12, 13, 14, 15);
}

// ---------------------------------------------------------------------------
// Kernel 3: 300-step LIF scan. 4 blocks x 256 threads; block owns batch rows
// m0..m0+15. Per wave (8 waves): layer1 -> 4 N-tiles, layer2 -> 2, readout -> 1.
// ---------------------------------------------------------------------------
__global__ __launch_bounds__(256) void snn_scan_kernel(
    const float* __restrict__ I1ff, const unsigned short* __restrict__ wrec,
    const unsigned short* __restrict__ w2b, const unsigned short* __restrict__ w3b,
    const float* __restrict__ alpha1, const float* __restrict__ rho1,
    const float* __restrict__ beta_a1, const float* __restrict__ alpha2,
    const float* __restrict__ rho2, const float* __restrict__ beta_a2,
    const float* __restrict__ beta_out, float* __restrict__ out) {
    __shared__ unsigned short s1b[2][16][512];  // 32 KB, double-buffered spikes L1
    __shared__ unsigned short s2b[2][16][256];  // 16 KB, double-buffered spikes L2

    const int lane = threadIdx.x & 31;
    const int wv   = threadIdx.x >> 5;
    const int lrow = lane & 15;
    const int lhi  = lane >> 4;
    const int m0   = blockIdx.x * 16;  // batch base

    for (int i = threadIdx.x; i < 2 * 16 * 512; i += 256) (&s1b[0][0][0])[i] = 0;
    for (int i = threadIdx.x; i < 2 * 16 * 256; i += 256) (&s2b[0][0][0])[i] = 0;
    __syncthreads();

    // --- per-neuron params (lane-invariant over the C-fragment's j dim) ---
    float al1[4], rh1[4], ba1[4];
#pragma unroll
    for (int ti = 0; ti < 4; ++ti) {
        int n = wv * 64 + ti * 16 + lrow;
        al1[ti] = alpha1[n]; rh1[ti] = rho1[n]; ba1[ti] = beta_a1[n];
    }
    float al2[2], rh2[2], ba2[2];
#pragma unroll
    for (int ti = 0; ti < 2; ++ti) {
        int n = wv * 32 + ti * 16 + lrow;
        al2[ti] = alpha2[n]; rh2[ti] = rho2[n]; ba2[ti] = beta_a2[n];
    }
    const int nout = wv * 16 + lrow;
    const float bo = (nout < 101) ? beta_out[nout] : 0.9f;

    // --- state in accumulator layout ---
    float v1[4][8] = {}, a1[4][8] = {};
    float v2[2][8] = {}, a2[2][8] = {};
    float vout[8] = {}, vsum[8] = {};

    int cur = 0;
#pragma unroll 1
    for (int t = 0; t < 300; ++t) {
        const int nxt = cur ^ 1;
        const float* I1t = I1ff + (size_t)t * 64 * 512;  // dense [64][512] slab

        // ================= layer 1: I1 = I1_ff[t] + s1_prev @ wrec^T =======
#pragma unroll 1
        for (int ti = 0; ti < 4; ++ti) {
            const int n = wv * 64 + ti * 16 + lrow;
            v8f acc;
#pragma unroll
            for (int j = 0; j < 8; ++j) {
                int b = m0 + j + (lhi << 3);
                acc[j] = I1t[(size_t)b * 512 + n];
            }
            const unsigned short* wr = wrec + (size_t)n * 512 + (lhi ? 16 : 0);
#pragma unroll 4
            for (int kb = 0; kb < 16; ++kb) {
                v16bf afrag = lds_afrag(&s1b[cur][0][0], lrow, lhi, kb, 512);
                v16bf bfrag = *(const v16bf*)(wr + kb * 32);
                acc = WMMA_BF16(afrag, bfrag, acc);
            }
            // adaptive-LIF update + soft reset
#pragma unroll
            for (int j = 0; j < 8; ++j) {
                int ml = j + (lhi << 3);
                float sprev = bf2f(s1b[cur][ml][n]);
                float vn = al1[ti] * v1[ti][j] + (1.f - al1[ti]) * (acc[j] - a1[ti][j]);
                float sn = (vn > 1.f) ? 1.f : 0.f;
                v1[ti][j] = vn - sn;
                a1[ti][j] = rh1[ti] * a1[ti][j] + ba1[ti] * sprev;
                s1b[nxt][ml][n] = f2bits(sn);
            }
        }
        __syncthreads();

        // ================= layer 2: I2 = s1_new @ w2^T ======================
#pragma unroll 1
        for (int ti = 0; ti < 2; ++ti) {
            const int n = wv * 32 + ti * 16 + lrow;
            v8f acc = {0.f, 0.f, 0.f, 0.f, 0.f, 0.f, 0.f, 0.f};
            const unsigned short* wr = w2b + (size_t)n * 512 + (lhi ? 16 : 0);
#pragma unroll 4
            for (int kb = 0; kb < 16; ++kb) {
                v16bf afrag = lds_afrag(&s1b[nxt][0][0], lrow, lhi, kb, 512);
                v16bf bfrag = *(const v16bf*)(wr + kb * 32);
                acc = WMMA_BF16(afrag, bfrag, acc);
            }
#pragma unroll
            for (int j = 0; j < 8; ++j) {
                int ml = j + (lhi << 3);
                float sprev = bf2f(s2b[cur][ml][n]);
                float vn = al2[ti] * v2[ti][j] + (1.f - al2[ti]) * (acc[j] - a2[ti][j]);
                float sn = (vn > 1.f) ? 1.f : 0.f;
                v2[ti][j] = vn - sn;
                a2[ti][j] = rh2[ti] * a2[ti][j] + ba2[ti] * sprev;
                s2b[nxt][ml][n] = f2bits(sn);
            }
        }
        __syncthreads();

        // ================= readout: I_out = s2_new @ w3^T (N padded to 128) =
        {
            v8f acc = {0.f, 0.f, 0.f, 0.f, 0.f, 0.f, 0.f, 0.f};
            const unsigned short* wr = w3b + (size_t)nout * 256 + (lhi ? 16 : 0);
#pragma unroll
            for (int kb = 0; kb < 8; ++kb) {
                v16bf afrag = lds_afrag(&s2b[nxt][0][0], lrow, lhi, kb, 256);
                v16bf bfrag = *(const v16bf*)(wr + kb * 32);
                acc = WMMA_BF16(afrag, bfrag, acc);
            }
#pragma unroll
            for (int j = 0; j < 8; ++j) {
                vout[j] = bo * vout[j] + (1.f - bo) * acc[j];
                vsum[j] += vout[j];
            }
        }
        __syncthreads();
        cur = nxt;
    }

    if (nout < 101) {
#pragma unroll
        for (int j = 0; j < 8; ++j) {
            int b = m0 + j + (lhi << 3);
            out[(size_t)b * 101 + nout] = vsum[j] * (1.f / 300.f);
        }
    }
}

// ---------------------------------------------------------------------------
// Launch: convert -> big GEMM -> scan (stream-ordered).
// Workspace: wrec_bf 512KB | w2_bf 256KB | w3_bf 64KB | I1_ff 37.5MB  (~40MB)
// ---------------------------------------------------------------------------
extern "C" void kernel_launch(void* const* d_in, const int* in_sizes, int n_in,
                              void* d_out, int out_size, void* d_ws, size_t ws_size,
                              hipStream_t stream) {
    const float* x       = (const float*)d_in[0];
    const float* w1      = (const float*)d_in[1];
    const float* wrec    = (const float*)d_in[2];
    const float* w2      = (const float*)d_in[3];
    const float* w3      = (const float*)d_in[4];
    const float* alpha1  = (const float*)d_in[5];
    const float* rho1    = (const float*)d_in[6];
    const float* beta_a1 = (const float*)d_in[7];
    const float* alpha2  = (const float*)d_in[8];
    const float* rho2    = (const float*)d_in[9];
    const float* beta_a2 = (const float*)d_in[10];
    const float* beta_out= (const float*)d_in[11];
    float* out = (float*)d_out;

    unsigned char* ws = (unsigned char*)d_ws;
    unsigned short* wrec_bf = (unsigned short*)(ws);
    unsigned short* w2_bf   = (unsigned short*)(ws + 524288);
    unsigned short* w3_bf   = (unsigned short*)(ws + 786432);
    float*          I1      = (float*)(ws + 851968);

    convert_weights_kernel<<<(512 * 512 + 256 * 512 + 128 * 256 + 255) / 256,
                             256, 0, stream>>>(wrec, w2, w3, wrec_bf, w2_bf, w3_bf);
    gemm_ff_kernel<<<dim3(300, 4), 256, 0, stream>>>(x, w1, I1);
    snn_scan_kernel<<<4, 256, 0, stream>>>(I1, wrec_bf, w2_bf, w3_bf,
                                           alpha1, rho1, beta_a1,
                                           alpha2, rho2, beta_a2,
                                           beta_out, out);
}
